// MultiQueryAttention_81449759801486
// MI455X (gfx1250) — matrix operand order repaired and checked by reading, not compiled, not run on hardware
//
#include <hip/hip_runtime.h>

#define DEVI __device__ __forceinline__

typedef __bf16 bf16_t;
typedef bf16_t v16bf __attribute__((ext_vector_type(16)));
typedef float  v8f   __attribute__((ext_vector_type(8)));
typedef unsigned int v4u __attribute__((ext_vector_type(4)));
typedef int v8i __attribute__((ext_vector_type(8)));
typedef int v4i __attribute__((ext_vector_type(4)));

union FragBF { v16bf v; uint4 q[2]; };

DEVI unsigned short f2bf_bits(float f) {
  unsigned u = __builtin_bit_cast(unsigned, f);
  u += 0x7FFFu + ((u >> 16) & 1u);          // round to nearest even
  return (unsigned short)(u >> 16);
}

DEVI v8f wmma_bf16(const FragBF& a, const FragBF& b, v8f c) {
  // (neg_a, A, neg_b, B, c_mod, C, reuse_a, reuse_b)
  return __builtin_amdgcn_wmma_f32_16x16x32_bf16(false, a.v, false, b.v,
                                                 (short)0, c, false, false);
}

// 16-lane butterfly via ds_swizzle (group-of-32, and=0x1F, xor in [14:10])
template <int IMM>
DEVI float swz(float x) {
  return __builtin_bit_cast(
      float, __builtin_amdgcn_ds_swizzle(__builtin_bit_cast(int, x), IMM));
}
#define SWZ_X1 0x041F
#define SWZ_X2 0x081F
#define SWZ_X4 0x101F
#define SWZ_X8 0x201F

// ---- Tensor Data Mover: 2D tile (tile_d1 rows x tile_d0 halves) -> LDS ----
DEVI void tdm_load_2d_bf16(unsigned ldsOff, const void* gaddr,
                           unsigned tensor_d0, unsigned tensor_d1,
                           unsigned tile_d0, unsigned tile_d1,
                           unsigned stride0) {
  unsigned long long ga = (unsigned long long)(size_t)gaddr;
  v4u g0;
  g0[0] = 1u;                                   // count=1, user descriptor
  g0[1] = ldsOff;                               // lds_addr (bytes)
  g0[2] = (unsigned)(ga & 0xFFFFFFFFu);         // global_addr[31:0]
  g0[3] = (unsigned)((ga >> 32) & 0x01FFFFFFu)  // global_addr[56:32]
          | (2u << 30);                         // type=2 ("image")
  v8i g1;
  g1[0] = 0x00010000;                           // wg_mask=0, data_size=1 (2B)
  g1[1] = (int)((tensor_d0 & 0xFFFFu) << 16);   // tensor_dim0[15:0] @ [63:48]
  g1[2] = (int)((tensor_d0 >> 16) | ((tensor_d1 & 0xFFFFu) << 16));
  g1[3] = (int)((tensor_d1 >> 16) | (tile_d0 << 16));  // tile_dim0 @ [127:112]
  g1[4] = (int)tile_d1;                         // tile_dim1; tile_dim2=0
  g1[5] = (int)stride0;                         // tensor_dim0_stride[31:0]
  g1[6] = 0;                                    // stride0 hi, dim1_stride lo
  g1[7] = 0;
  v4i z4 = {};
#if __clang_major__ >= 23
  v8i z8 = {};
  __builtin_amdgcn_tensor_load_to_lds(g0, g1, z4, z4, z8, 0);
#else
  __builtin_amdgcn_tensor_load_to_lds(g0, g1, z4, z4, 0);
#endif
}

// ---------------------------------------------------------------- cast kernels
__global__ void mqa_cast_f32_bf16(const float* __restrict__ x,
                                  unsigned short* __restrict__ y, int n) {
  int i = blockIdx.x * blockDim.x + threadIdx.x;
  if (i < n) y[i] = f2bf_bits(x[i]);
}

// W [K,N] fp32 -> WT [N,K] bf16  (so WMMA B-fragments are contiguous per lane)
__global__ void mqa_cast_w_transpose(const float* __restrict__ w,
                                     unsigned short* __restrict__ wt,
                                     int K, int N) {
  int i = blockIdx.x * blockDim.x + threadIdx.x;
  if (i < K * N) {
    int k = i / N, n = i % N;
    wt[(size_t)n * K + k] = f2bf_bits(w[i]);
  }
}

// V [B*T, D] bf16 -> VT [B, D, T] bf16
__global__ void mqa_transpose_v(const unsigned short* __restrict__ v,
                                unsigned short* __restrict__ vt,
                                int Bb, int T, int D) {
  int i = blockIdx.x * blockDim.x + threadIdx.x;
  if (i < Bb * T * D) {
    int bt = i / D, d = i % D;
    int b = bt / T, t = bt % T;
    vt[((size_t)b * D + d) * T + t] = v[i];
  }
}

// ---------------------------------------------------------------- WMMA GEMM
// C[M,N] = A[M,K](bf16,row-major) * BT[N,K](bf16)^T + bias[N]
// Block: 256 threads = 8 waves. Wave w -> rows [by*128 + 16w, +16), cols [bx*64, +64).
// The shared 64xKC B tile is staged into LDS once per block by the TDM.
template <bool F32OUT>
__global__ void mqa_gemm_wmma(const unsigned short* __restrict__ A,
                              const unsigned short* __restrict__ BT,
                              const float* __restrict__ bias,
                              void* __restrict__ Out,
                              int M, int N, int K) {
  constexpr int KC = 64;                            // K-chunk per TDM stage
  __shared__ __align__(16) unsigned short bTile[64][KC];  // 8 KB

  const int lane = threadIdx.x & 31;
  const int wid  = threadIdx.x >> 5;
  const int half = lane >> 4;        // 0: lanes 0-15, 1: lanes 16-31
  const int l16  = lane & 15;
  const int m0 = blockIdx.y * 128 + wid * 16;
  const int n0 = blockIdx.x * 64;

  const uint4* arow = (const uint4*)(A + (size_t)(m0 + l16) * K);
  const unsigned short* bBase = BT + (size_t)n0 * K;
  const unsigned ldsOff = (unsigned)(size_t)&bTile[0][0];

  v8f acc[4];
#pragma unroll
  for (int j = 0; j < 4; ++j) acc[j] = (v8f){};

  for (int k0 = 0; k0 < K; k0 += KC) {
    __syncthreads();                 // previous tile fully consumed
    if (wid == 0) {
      tdm_load_2d_bf16(ldsOff, bBase + k0,
                       /*tensor_d0=*/K, /*tensor_d1=*/64,
                       /*tile_d0=*/KC, /*tile_d1=*/64, /*stride0=*/K);
      __builtin_amdgcn_s_wait_tensorcnt(0);
    }
    __syncthreads();                 // tile visible to all waves

#pragma unroll
    for (int ks = 0; ks < 2; ++ks) { // two 32-deep WMMA steps per stage
      const int kk = k0 + 32 * ks;
      FragBF a;                      // 16x32 A-fragment (contiguous k = 16*half + s)
      a.q[0] = arow[(kk >> 3) + 2 * half];
      a.q[1] = arow[(kk >> 3) + 2 * half + 1];
#pragma unroll
      for (int j = 0; j < 4; ++j) {  // 32x16 B-fragment from LDS: lane = col n
        const uint4* brow = (const uint4*)&bTile[16 * j + l16][32 * ks + 16 * half];
        FragBF b;
        b.q[0] = brow[0];
        b.q[1] = brow[1];
        acc[j] = wmma_bf16(a, b, acc[j]);
      }
    }
  }

  const int col = n0 + l16;
#pragma unroll
  for (int j = 0; j < 4; ++j) {
    const float bj = bias[col + 16 * j];
#pragma unroll
    for (int r = 0; r < 8; ++r) {
      const int orow = m0 + r + 8 * half;        // C layout: vgpr r, half-> +8
      const float val = acc[j][r] + bj;
      if (F32OUT)
        ((float*)Out)[(size_t)orow * N + col + 16 * j] = val;
      else
        ((unsigned short*)Out)[(size_t)orow * N + col + 16 * j] = f2bf_bits(val);
    }
  }
}

// ---------------------------------------------------------------- flash MQA
// One wave = one 16-row Q tile of one (b,h). Streams K/V in 32-key chunks.
__global__ void mqa_flash_wmma(const unsigned short* __restrict__ Q,   // [B,T,H*D] bf16
                               const unsigned short* __restrict__ Kb,  // [B,T,D]   bf16
                               const unsigned short* __restrict__ VT,  // [B,D,T]   bf16
                               unsigned short* __restrict__ O) {       // [B,T,H*D] bf16
  constexpr int T = 2048, H = 16, D = 64;
  __shared__ __align__(16) unsigned short pTile[8][16][32];  // per-wave P tile

  const int lane = threadIdx.x & 31;
  const int wid  = threadIdx.x >> 5;
  const int half = lane >> 4;
  const int l16  = lane & 15;

  const int g  = blockIdx.x * 8 + wid;           // global wave id
  const int b  = g / (H * (T / 16));
  const int rm = g % (H * (T / 16));
  const int h  = rm / (T / 16);
  const int q0 = (rm % (T / 16)) * 16;

  // Q A-fragments over d: qa0 = d[0..31], qa1 = d[32..63]
  const uint4* qrow =
      (const uint4*)(Q + (size_t)(b * T + q0 + l16) * (H * D) + h * D);
  FragBF qa0, qa1;
  qa0.q[0] = qrow[2 * half];     qa0.q[1] = qrow[2 * half + 1];
  qa1.q[0] = qrow[4 + 2 * half]; qa1.q[1] = qrow[5 + 2 * half];

  const unsigned short* kbase = Kb + (size_t)b * T * D;
  const unsigned short* vtb   = VT + (size_t)b * D * T;
  const uint4* vrow[4];
#pragma unroll
  for (int t = 0; t < 4; ++t)
    vrow[t] = (const uint4*)(vtb + (size_t)(16 * t + l16) * T);

  v8f o0 = (v8f){}, o1 = (v8f){}, o2 = (v8f){}, o3 = (v8f){};
  float m[8], lsum[8];
#pragma unroll
  for (int r = 0; r < 8; ++r) { m[r] = -1e30f; lsum[r] = 0.f; }

  constexpr float SC = 0.125f * 1.44269504f;     // 1/sqrt(64) * log2(e)

  for (int kc = 0; kc < T; kc += 32) {
    // ---- S = Q * K^T  (two 16x16 key tiles, K-dim = 64 over 2 wmma each)
    const uint4* kr0 = (const uint4*)(kbase + (size_t)(kc + l16) * D);
    const uint4* kr1 = (const uint4*)(kbase + (size_t)(kc + 16 + l16) * D);
    if (kc + 32 < T)
      __builtin_prefetch(kbase + (size_t)(kc + 32 + l16) * D, 0, 0);

    FragBF kf;
    v8f s0 = (v8f){}, s1 = (v8f){};
    kf.q[0] = kr0[2 * half];     kf.q[1] = kr0[2 * half + 1];
    s0 = wmma_bf16(qa0, kf, s0);
    kf.q[0] = kr0[4 + 2 * half]; kf.q[1] = kr0[4 + 2 * half + 1];
    s0 = wmma_bf16(qa1, kf, s0);
    kf.q[0] = kr1[2 * half];     kf.q[1] = kr1[2 * half + 1];
    s1 = wmma_bf16(qa0, kf, s1);
    kf.q[0] = kr1[4 + 2 * half]; kf.q[1] = kr1[4 + 2 * half + 1];
    s1 = wmma_bf16(qa1, kf, s1);

    // ---- online softmax (exp2 domain); row = vgpr index, cols across 16 lanes
    float p0[8], p1[8], alpha[8];
#pragma unroll
    for (int r = 0; r < 8; ++r) {
      float a0 = s0[r] * SC, a1 = s1[r] * SC;
      float mx = fmaxf(a0, a1);
      mx = fmaxf(mx, swz<SWZ_X1>(mx));
      mx = fmaxf(mx, swz<SWZ_X2>(mx));
      mx = fmaxf(mx, swz<SWZ_X4>(mx));
      mx = fmaxf(mx, swz<SWZ_X8>(mx));
      const float mn = fmaxf(m[r], mx);
      const float al = exp2f(m[r] - mn);
      m[r] = mn;
      p0[r] = exp2f(a0 - mn);
      p1[r] = exp2f(a1 - mn);
      lsum[r] = lsum[r] * al + p0[r] + p1[r];
      alpha[r] = al;
    }
#pragma unroll
    for (int r = 0; r < 8; ++r) {
      o0[r] *= alpha[r]; o1[r] *= alpha[r];
      o2[r] *= alpha[r]; o3[r] *= alpha[r];
    }

    // ---- C-layout P -> LDS -> A-layout P fragment (wave-private tile)
#pragma unroll
    for (int r = 0; r < 8; ++r) {
      pTile[wid][r + 8 * half][l16]      = f2bf_bits(p0[r]);
      pTile[wid][r + 8 * half][16 + l16] = f2bf_bits(p1[r]);
    }
    asm volatile("s_wait_dscnt 0" ::: "memory");
    FragBF pf;
    const uint4* prow = (const uint4*)&pTile[wid][l16][0];
    pf.q[0] = prow[2 * half];
    pf.q[1] = prow[2 * half + 1];

    // ---- O += P * V  (4 d-tiles of 16, K-dim = 32 keys)
    const int vi = (kc >> 3) + 2 * half;
    FragBF vf;
    vf.q[0] = vrow[0][vi]; vf.q[1] = vrow[0][vi + 1];
    o0 = wmma_bf16(pf, vf, o0);
    vf.q[0] = vrow[1][vi]; vf.q[1] = vrow[1][vi + 1];
    o1 = wmma_bf16(pf, vf, o1);
    vf.q[0] = vrow[2][vi]; vf.q[1] = vrow[2][vi + 1];
    o2 = wmma_bf16(pf, vf, o2);
    vf.q[0] = vrow[3][vi]; vf.q[1] = vrow[3][vi + 1];
    o3 = wmma_bf16(pf, vf, o3);
  }

  // ---- finalize: divide by row sums, emit bf16 for the output projection
#pragma unroll
  for (int r = 0; r < 8; ++r) {
    float l = lsum[r];
    l += swz<SWZ_X1>(l);
    l += swz<SWZ_X2>(l);
    l += swz<SWZ_X4>(l);
    l += swz<SWZ_X8>(l);
    const float inv = 1.f / l;
    const int orow = q0 + r + 8 * half;
    const size_t base = (size_t)(b * T + orow) * (H * D) + h * D + l16;
    O[base]      = f2bf_bits(o0[r] * inv);
    O[base + 16] = f2bf_bits(o1[r] * inv);
    O[base + 32] = f2bf_bits(o2[r] * inv);
    O[base + 48] = f2bf_bits(o3[r] * inv);
  }
}

// ---------------------------------------------------------------- launch
extern "C" void kernel_launch(void* const* d_in, const int* in_sizes, int n_in,
                              void* d_out, int out_size, void* d_ws, size_t ws_size,
                              hipStream_t stream) {
  const float* query = (const float*)d_in[0];
  const float* Wq    = (const float*)d_in[1];
  const float* bq    = (const float*)d_in[2];
  const float* Wk    = (const float*)d_in[3];
  const float* bk    = (const float*)d_in[4];
  const float* Wv    = (const float*)d_in[5];
  const float* bv    = (const float*)d_in[6];
  const float* Wo    = (const float*)d_in[7];
  const float* bo    = (const float*)d_in[8];
  float* out = (float*)d_out;

  constexpr int Bb = 2, T = 2048, DM = 1024, H = 16, D = 64;
  constexpr int M = Bb * T;  // 4096

  char* ws = (char*)d_ws;
  size_t off = 0;
  auto alloc = [&](size_t bytes) {
    void* p = ws + off;
    off = (off + bytes + 255) & ~(size_t)255;
    return p;
  };
  unsigned short* xbf  = (unsigned short*)alloc((size_t)M * DM * 2);
  unsigned short* WqT  = (unsigned short*)alloc((size_t)DM * DM * 2);
  unsigned short* WkT  = (unsigned short*)alloc((size_t)D * DM * 2);
  unsigned short* WvT  = (unsigned short*)alloc((size_t)D * DM * 2);
  unsigned short* WoT  = (unsigned short*)alloc((size_t)DM * DM * 2);
  unsigned short* qbf  = (unsigned short*)alloc((size_t)M * DM * 2);
  unsigned short* kbf  = (unsigned short*)alloc((size_t)M * D * 2);
  unsigned short* vbf  = (unsigned short*)alloc((size_t)M * D * 2);
  unsigned short* vT   = (unsigned short*)alloc((size_t)M * D * 2);
  unsigned short* attn = (unsigned short*)alloc((size_t)M * DM * 2);

  const int n1 = M * DM;
  mqa_cast_f32_bf16<<<(n1 + 255) / 256, 256, 0, stream>>>(query, xbf, n1);
  mqa_cast_w_transpose<<<(DM * DM + 255) / 256, 256, 0, stream>>>(Wq, WqT, DM, DM);
  mqa_cast_w_transpose<<<(DM * D  + 255) / 256, 256, 0, stream>>>(Wk, WkT, DM, D);
  mqa_cast_w_transpose<<<(DM * D  + 255) / 256, 256, 0, stream>>>(Wv, WvT, DM, D);
  mqa_cast_w_transpose<<<(DM * DM + 255) / 256, 256, 0, stream>>>(Wo, WoT, DM, DM);

  mqa_gemm_wmma<false><<<dim3(DM / 64, M / 128), 256, 0, stream>>>(
      xbf, WqT, bq, qbf, M, DM, DM);
  mqa_gemm_wmma<false><<<dim3(D / 64, M / 128), 256, 0, stream>>>(
      xbf, WkT, bk, kbf, M, D, DM);
  mqa_gemm_wmma<false><<<dim3(D / 64, M / 128), 256, 0, stream>>>(
      xbf, WvT, bv, vbf, M, D, DM);

  mqa_transpose_v<<<(M * D + 255) / 256, 256, 0, stream>>>(vbf, vT, Bb, T, D);

  mqa_flash_wmma<<<(Bb * H * (T / 16)) / 8, 256, 0, stream>>>(qbf, kbf, vT, attn);

  mqa_gemm_wmma<true><<<dim3(DM / 64, M / 128), 256, 0, stream>>>(
      attn, WoT, bo, out, M, DM, DM);
}
